// CrossAttn_60687887893118
// MI455X (gfx1250) — matrix operand order repaired and checked by reading, not compiled, not run on hardware
//
#include <hip/hip_runtime.h>
#include <hip/hip_bf16.h>
#include <math.h>

// ---------------- problem constants (match reference) ----------------
#define B_    2
#define HW_   48               // H = W = 48 feature map
#define DIM_  768
#define NH_   6
#define NP_   4
#define DH_   128              // DIM/NH
#define HID_  192              // DIM/4
#define LIN_  (HW_ * HW_)      // 2304
#define LQ_   (21 * (HW_ * HW_) / 4)   // 12096
#define MQ_   (B_ * LQ_)       // 24192  (divisible by 64)
#define MV_   (B_ * LIN_)      // 4608   (divisible by 64)
#define NPAD_ 64               // padded N for the two small GEMMs

typedef __bf16 bf16_t;
typedef bf16_t v16bf __attribute__((ext_vector_type(16)));
typedef bf16_t v8bf  __attribute__((ext_vector_type(8)));
typedef float  v8f   __attribute__((ext_vector_type(8)));
typedef int    v4i   __attribute__((vector_size(16)));
typedef __attribute__((address_space(1))) v4i* gas_v4i_ptr;
typedef __attribute__((address_space(3))) v4i* las_v4i_ptr;

#if __has_builtin(__builtin_amdgcn_global_load_async_to_lds_b128)
#define HAS_ASYNC_LDS 1
#else
#define HAS_ASYNC_LDS 0
#endif

static __device__ __forceinline__ void wait_async0() {
#if __has_builtin(__builtin_amdgcn_s_wait_asynccnt)
  __builtin_amdgcn_s_wait_asynccnt(0);
#else
  asm volatile("s_wait_asynccnt 0x0" ::: "memory");
#endif
}

#if HAS_ASYNC_LDS
static __device__ __forceinline__ void stage_async_b128(const bf16_t* g, bf16_t* l) {
  __builtin_amdgcn_global_load_async_to_lds_b128(
      (gas_v4i_ptr)(g), (las_v4i_ptr)(l), 0, 0);
}
#endif

static __device__ __forceinline__ bf16_t f2bf(float f) {
  unsigned u = __builtin_bit_cast(unsigned, f);
  u += 0x7FFFu + ((u >> 16) & 1u);               // round-to-nearest-even
  unsigned short h = (unsigned short)(u >> 16);
  return __builtin_bit_cast(bf16_t, h);
}

// ---------------- fp32 -> bf16 convert ----------------
__global__ __launch_bounds__(256) void cvt_f32_bf16(const float* __restrict__ s,
                                                    bf16_t* __restrict__ d, int n) {
  int i = blockIdx.x * 256 + threadIdx.x;
  if (i < n) d[i] = f2bf(s[i]);
}

// convert [K,N] -> [K,Npad] with zero padding of extra columns
__global__ __launch_bounds__(256) void cvt_f32_bf16_pad(const float* __restrict__ s,
                                                        bf16_t* __restrict__ d,
                                                        int K, int N, int Npad) {
  int i = blockIdx.x * 256 + threadIdx.x;
  if (i >= K * Npad) return;
  int row = i / Npad, col = i % Npad;
  d[i] = (col < N) ? f2bf(s[row * N + col]) : __builtin_bit_cast(bf16_t, (unsigned short)0);
}

__global__ __launch_bounds__(64) void pad_bias_f32(const float* __restrict__ s,
                                                   float* __restrict__ d, int N, int Npad) {
  int i = threadIdx.x;
  if (i < Npad) d[i] = (i < N) ? s[i] : 0.f;
}

// ---------------- LayerNorm(row of 768) -> bf16 ----------------
__global__ __launch_bounds__(256) void ln_to_bf16(const float* __restrict__ in,
                                                  const float* __restrict__ g,
                                                  const float* __restrict__ bta,
                                                  bf16_t* __restrict__ out, int cols) {
  __shared__ float red[256];
  const int row = blockIdx.x;
  const float* x = in + (size_t)row * cols;
  float s = 0.f;
  for (int i = threadIdx.x; i < cols; i += 256) s += x[i];
  red[threadIdx.x] = s; __syncthreads();
  for (int st = 128; st > 0; st >>= 1) {
    if (threadIdx.x < st) red[threadIdx.x] += red[threadIdx.x + st];
    __syncthreads();
  }
  const float mean = red[0] / (float)cols;
  __syncthreads();
  float s2 = 0.f;
  for (int i = threadIdx.x; i < cols; i += 256) { float d = x[i] - mean; s2 += d * d; }
  red[threadIdx.x] = s2; __syncthreads();
  for (int st = 128; st > 0; st >>= 1) {
    if (threadIdx.x < st) red[threadIdx.x] += red[threadIdx.x + st];
    __syncthreads();
  }
  const float rstd = rsqrtf(red[0] / (float)cols + 1e-6f);
  for (int i = threadIdx.x; i < cols; i += 256)
    out[(size_t)row * cols + i] = f2bf((x[i] - mean) * rstd * g[i] + bta[i]);
}

// ---------------- bf16 WMMA GEMM (guard-free) ----------------
// C(f32)[M,N] = A(bf16)[M,K] @ B(bf16)[K,N] + bias (+resid)
// Requirements: M%64==0, N%64==0, K%32==0, K>=64.
// Block = 256 threads = 8 waves arranged 4(M) x 2(N); block tile 64x64.
// B slab (32 x 64) double-buffered in LDS via async global->LDS DMA (ASYNCcnt),
// overlapping next-slab DMA with current-slab WMMA. Last k-step peeled so the
// steady-state loop has no branches. A fragments direct from global.
__global__ __launch_bounds__(256) void gemm_bf16_wmma(
    const bf16_t* __restrict__ A, const bf16_t* __restrict__ Bm,
    const float* __restrict__ bias, const float* __restrict__ resid,
    float* __restrict__ C, int M, int N, int K) {
  const int lane = threadIdx.x & 31;
  const int wave = threadIdx.x >> 5;
  const int wm   = wave & 3;           // M subtile 0..3
  const int wn   = wave >> 2;          // N half    0..1
  const int m0   = blockIdx.y * 64 + wm * 16;
  const int nB   = blockIdx.x * 64;    // block N base
  const int n0   = nB + wn * 32;       // wave N base

  const int mRow = m0 + (lane & 15);
  const int kSel = (lane < 16) ? 0 : 8;          // A-layout K select per lane half

  // B staging coords: each thread moves 8 bf16 (16B) of the 32x64 slab
  const int gRow = threadIdx.x >> 3;             // 0..31
  const int gCol = (threadIdx.x & 7) * 8;        // 0,8,..,56
  const bf16_t* gstage = Bm + (size_t)gRow * N + nB + gCol;   // advance by 32*N per slab

  v8f acc0 = {}; v8f acc1 = {};

  auto compute = [&](int kk, const bf16_t* bufL) {
    // A fragment: lane holds row (lane&15), K = {kSel..+7, kSel+16..+23}
    const bf16_t* ap = A + (size_t)mRow * K + (kk + kSel);
    v8bf alo = *(const v8bf*)ap;
    v8bf ahi = *(const v8bf*)(ap + 16);
    v16bf a = __builtin_shufflevector(alo, ahi, 0,1,2,3,4,5,6,7,8,9,10,11,12,13,14,15);

    const bf16_t* bsp = bufL + wn * 32 + (lane & 15);
    v16bf b0, b1;
    #pragma unroll
    for (int s = 0; s < 16; ++s) {
      const int kb = (s < 8) ? (kSel + s) : (kSel + 8 + s);
      b0[s] = bsp[kb * 64];
      b1[s] = bsp[kb * 64 + 16];
    }

    acc0 = __builtin_amdgcn_wmma_f32_16x16x32_bf16(false, a, false, b0,
                                                   (short)0, acc0, false, false);
    acc1 = __builtin_amdgcn_wmma_f32_16x16x32_bf16(false, a, false, b1,
                                                   (short)0, acc1, false, false);
  };

#if HAS_ASYNC_LDS
  __shared__ bf16_t Bs[2][32 * 64];
  const int lidx = gRow * 64 + gCol;
  bf16_t* curL = &Bs[0][0];
  bf16_t* nxtL = &Bs[1][0];

  // prologue: async DMA of slab 0 into buffer 0
  stage_async_b128(gstage, curL + lidx);

  int k0 = 0;
  for (; k0 < K - 32; k0 += 32) {
    wait_async0();                               // my current-slab DMA complete
    __syncthreads();                             // all waves' DMAs complete; prev reads retired
    stage_async_b128(gstage + (size_t)(k0 + 32) * N, nxtL + lidx);  // overlap with compute
    compute(k0, curL);
    bf16_t* t = curL; curL = nxtL; nxtL = t;
  }
  // peeled last slab: nothing left to stage
  wait_async0();
  __syncthreads();
  compute(k0, curL);
#else
  __shared__ bf16_t Bs[32 * 64];
  for (int k0 = 0; k0 < K; k0 += 32) {
    __syncthreads();                             // protect LDS from previous reads
    *(v8bf*)(Bs + gRow * 64 + gCol) =
        *(const v8bf*)(gstage + (size_t)k0 * N);
    __syncthreads();
    compute(k0, Bs);
  }
#endif

  // epilogue: lane L, reg r -> M = r + (L>=16?8:0), N = L&15
  const int col0 = n0 + (lane & 15);
  const int rowB = m0 + ((lane >> 4) << 3);
  const float bia0 = bias[col0];
  const float bia1 = bias[col0 + 16];
  #pragma unroll
  for (int r = 0; r < 8; ++r) {
    const size_t mrow = (size_t)(rowB + r) * (size_t)N;
    float v0 = acc0[r] + bia0;
    float v1 = acc1[r] + bia1;
    if (resid) { v0 += resid[mrow + col0]; v1 += resid[mrow + col0 + 16]; }
    C[mrow + col0]      = v0;
    C[mrow + col0 + 16] = v1;
  }
}

// ---------------- deformable sampling: softmax(attw) . bilinear(value) -> bf16 ----------------
__global__ __launch_bounds__(128) void deform_sample(
    const float* __restrict__ val,   // [B, LIN, DIM] (channel = h*DH + d)
    const float* __restrict__ refp,  // [B*LQ, 2]
    const float* __restrict__ offs,  // [B*LQ, NPAD] (valid cols: 48)
    const float* __restrict__ attl,  // [B*LQ, NPAD] (valid cols: 24) logits
    bf16_t* __restrict__ out) {      // [B*LQ, DIM]
  const int h  = blockIdx.x % NH_;
  const int bq = blockIdx.x / NH_;
  const int b  = bq / LQ_;
  const int d  = threadIdx.x;

  const float* o  = offs + (size_t)bq * NPAD_ + h * (NP_ * 2);
  const float* al = attl + (size_t)bq * NPAD_ + h * NP_;
  const float l0 = al[0], l1 = al[1], l2 = al[2], l3 = al[3];
  const float mx = fmaxf(fmaxf(l0, l1), fmaxf(l2, l3));
  const float e0 = expf(l0 - mx), e1 = expf(l1 - mx), e2 = expf(l2 - mx), e3 = expf(l3 - mx);
  const float inv = 1.f / (e0 + e1 + e2 + e3);
  const float w[4] = { e0 * inv, e1 * inv, e2 * inv, e3 * inv };

  const float rx = refp[(size_t)bq * 2 + 0] * (float)HW_;
  const float ry = refp[(size_t)bq * 2 + 1] * (float)HW_;
  const float* vb = val + (size_t)b * LIN_ * DIM_ + h * DH_ + d;

  float acc = 0.f;
  #pragma unroll
  for (int p = 0; p < NP_; ++p) {
    const float px = rx + o[p * 2 + 0] - 0.5f;     // loc*W - 0.5, align_corners=False
    const float py = ry + o[p * 2 + 1] - 0.5f;
    const float fx = floorf(px), fy = floorf(py);
    const float wx = px - fx, wy = py - fy;
    const int x0 = (int)fx, y0 = (int)fy;
    float s = 0.f;
    #pragma unroll
    for (int cy = 0; cy < 2; ++cy) {
      #pragma unroll
      for (int cx = 0; cx < 2; ++cx) {
        const int xi = x0 + cx, yi = y0 + cy;
        const float wgt = (cy ? wy : 1.f - wy) * (cx ? wx : 1.f - wx);
        if (xi >= 0 && xi < HW_ && yi >= 0 && yi < HW_)
          s += wgt * vb[(size_t)(yi * HW_ + xi) * DIM_];
      }
    }
    acc += w[p] * s;
  }
  out[(size_t)bq * DIM_ + h * DH_ + d] = f2bf(acc);
}

// ---------------- depthwise 3x3 (3 scale segments) + exact GELU -> bf16 ----------------
__global__ __launch_bounds__(256) void dwconv_gelu(
    const float* __restrict__ hin,   // [B, LQ, HID]
    const float* __restrict__ w,     // [HID, 9]
    const float* __restrict__ bias,  // [HID]
    bf16_t* __restrict__ out) {      // [B, LQ, HID]
  const int tid = blockIdx.x * 256 + threadIdx.x;
  const int c = tid % HID_;
  const int t = (tid / HID_) % LQ_;
  const int b = tid / (HID_ * LQ_);
  int Hd, Wd, base;
  if (t < 16 * (LQ_ / 21))      { Hd = 2 * HW_; Wd = 2 * HW_; base = 0; }
  else if (t < 20 * (LQ_ / 21)) { Hd = HW_;     Wd = HW_;     base = 16 * (LQ_ / 21); }
  else                          { Hd = HW_ / 2; Wd = HW_ / 2; base = 20 * (LQ_ / 21); }
  const int local = t - base;
  const int rr = local / Wd, cc = local % Wd;
  const float* hb = hin + (size_t)b * LQ_ * HID_;
  float acc = bias[c];
  #pragma unroll
  for (int ky = -1; ky <= 1; ++ky) {
    #pragma unroll
    for (int kx = -1; kx <= 1; ++kx) {
      const int r2 = rr + ky, c2 = cc + kx;
      if (r2 >= 0 && r2 < Hd && c2 >= 0 && c2 < Wd)
        acc += hb[(size_t)(base + r2 * Wd + c2) * HID_ + c] * w[c * 9 + (ky + 1) * 3 + (kx + 1)];
    }
  }
  const float g = 0.5f * acc * (1.f + erff(acc * 0.70710678118654752f));
  out[(size_t)tid] = f2bf(g);
}

// ---------------- host orchestration ----------------
extern "C" void kernel_launch(void* const* d_in, const int* in_sizes, int n_in,
                              void* d_out, int out_size, void* d_ws, size_t ws_size,
                              hipStream_t stream) {
  (void)in_sizes; (void)n_in; (void)out_size; (void)ws_size;
  const float* query = (const float*)d_in[0];
  const float* refp  = (const float*)d_in[1];
  const float* feat  = (const float*)d_in[2];
  // d_in[3..6]: spatial_shapes, level_start_index, H, W -> static here
  const float* qn_g = (const float*)d_in[7];
  const float* qn_b = (const float*)d_in[8];
  const float* fn_g = (const float*)d_in[9];
  const float* fn_b = (const float*)d_in[10];
  const float* mn_g = (const float*)d_in[11];
  const float* mn_b = (const float*)d_in[12];
  const float* vW   = (const float*)d_in[13];
  const float* vb   = (const float*)d_in[14];
  const float* soW  = (const float*)d_in[15];
  const float* sob  = (const float*)d_in[16];
  const float* awW  = (const float*)d_in[17];
  const float* awb  = (const float*)d_in[18];
  const float* opW  = (const float*)d_in[19];
  const float* opb  = (const float*)d_in[20];
  const float* fc1W = (const float*)d_in[21];
  const float* fc1b = (const float*)d_in[22];
  const float* dwW  = (const float*)d_in[23];
  const float* dwb  = (const float*)d_in[24];
  const float* fc2W = (const float*)d_in[25];
  const float* fc2b = (const float*)d_in[26];

  char* ws = (char*)d_ws;
  size_t off = 0;
  auto alloc = [&](size_t bytes) -> void* {
    bytes = (bytes + 255) & ~(size_t)255;
    void* p = ws + off; off += bytes; return p;
  };

  bf16_t* w_v   = (bf16_t*)alloc((size_t)DIM_ * DIM_ * 2);
  bf16_t* w_so  = (bf16_t*)alloc((size_t)DIM_ * NPAD_ * 2);   // padded 48->64
  bf16_t* w_aw  = (bf16_t*)alloc((size_t)DIM_ * NPAD_ * 2);   // padded 24->64
  bf16_t* w_op  = (bf16_t*)alloc((size_t)DIM_ * DIM_ * 2);
  bf16_t* w_fc1 = (bf16_t*)alloc((size_t)DIM_ * HID_ * 2);
  bf16_t* w_fc2 = (bf16_t*)alloc((size_t)HID_ * DIM_ * 2);
  float*  sob_p = (float*) alloc(NPAD_ * 4);
  float*  awb_p = (float*) alloc(NPAD_ * 4);
  bf16_t* fln   = (bf16_t*)alloc((size_t)MV_ * DIM_ * 2);
  float*  vval  = (float*) alloc((size_t)MV_ * DIM_ * 4);
  bf16_t* qln   = (bf16_t*)alloc((size_t)MQ_ * DIM_ * 2);   // reused later as LN(x)
  float*  offsb = (float*) alloc((size_t)MQ_ * NPAD_ * 4);
  float*  attlb = (float*) alloc((size_t)MQ_ * NPAD_ * 4);
  bf16_t* attn  = (bf16_t*)alloc((size_t)MQ_ * DIM_ * 2);
  float*  xbuf  = (float*) alloc((size_t)MQ_ * DIM_ * 4);
  float*  hbuf  = (float*) alloc((size_t)MQ_ * HID_ * 4);
  bf16_t* hg    = (bf16_t*)alloc((size_t)MQ_ * HID_ * 2);

  auto cvt = [&](const float* s, bf16_t* d, int n) {
    cvt_f32_bf16<<<(n + 255) / 256, 256, 0, stream>>>(s, d, n);
  };
  cvt(vW,   w_v,   DIM_ * DIM_);
  cvt(opW,  w_op,  DIM_ * DIM_);
  cvt(fc1W, w_fc1, DIM_ * HID_);
  cvt(fc2W, w_fc2, HID_ * DIM_);
  cvt_f32_bf16_pad<<<(DIM_ * NPAD_ + 255) / 256, 256, 0, stream>>>(
      soW, w_so, DIM_, NH_ * NP_ * 2, NPAD_);
  cvt_f32_bf16_pad<<<(DIM_ * NPAD_ + 255) / 256, 256, 0, stream>>>(
      awW, w_aw, DIM_, NH_ * NP_, NPAD_);
  pad_bias_f32<<<1, 64, 0, stream>>>(sob, sob_p, NH_ * NP_ * 2, NPAD_);
  pad_bias_f32<<<1, 64, 0, stream>>>(awb, awb_p, NH_ * NP_, NPAD_);

  // LN(feat), LN(query)
  ln_to_bf16<<<MV_, 256, 0, stream>>>(feat,  fn_g, fn_b, fln, DIM_);
  ln_to_bf16<<<MQ_, 256, 0, stream>>>(query, qn_g, qn_b, qln, DIM_);

  // value = LN(feat) @ vW + vb                      [MV, 768]
  gemm_bf16_wmma<<<dim3(DIM_ / 64, MV_ / 64), 256, 0, stream>>>(
      fln, w_v, vb, nullptr, vval, MV_, DIM_, DIM_);
  // sampling offsets = qln @ soW + sob              [MQ, 64(pad)]
  gemm_bf16_wmma<<<dim3(1, MQ_ / 64), 256, 0, stream>>>(
      qln, w_so, sob_p, nullptr, offsb, MQ_, NPAD_, DIM_);
  // attention logits = qln @ awW + awb              [MQ, 64(pad)]
  gemm_bf16_wmma<<<dim3(1, MQ_ / 64), 256, 0, stream>>>(
      qln, w_aw, awb_p, nullptr, attlb, MQ_, NPAD_, DIM_);

  // softmax + bilinear gather + head merge -> attn (bf16)
  deform_sample<<<(size_t)B_ * LQ_ * NH_, 128, 0, stream>>>(vval, refp, offsb, attlb, attn);

  // x = query + attn @ opW + opb                    [MQ, 768]
  gemm_bf16_wmma<<<dim3(DIM_ / 64, MQ_ / 64), 256, 0, stream>>>(
      attn, w_op, opb, query, xbuf, MQ_, DIM_, DIM_);

  // LN(x) -> qln (buffer reuse)
  ln_to_bf16<<<MQ_, 256, 0, stream>>>(xbuf, mn_g, mn_b, qln, DIM_);

  // h = LN(x) @ fc1W + fc1b                         [MQ, 192]
  gemm_bf16_wmma<<<dim3(HID_ / 64, MQ_ / 64), 256, 0, stream>>>(
      qln, w_fc1, fc1b, nullptr, hbuf, MQ_, HID_, DIM_);

  // depthwise 3x3 per scale + GELU -> hg (bf16)
  dwconv_gelu<<<(B_ * LQ_ * HID_) / 256, 256, 0, stream>>>(hbuf, dwW, dwb, hg);

  // out = x + hg @ fc2W + fc2b                      [MQ, 768]
  gemm_bf16_wmma<<<dim3(DIM_ / 64, MQ_ / 64), 256, 0, stream>>>(
      hg, w_fc2, fc2b, xbuf, (float*)d_out, MQ_, DIM_, HID_);
}
